// Attention_65283502899297
// MI455X (gfx1250) — matrix operand order repaired and checked by reading, not compiled, not run on hardware
//
#include <hip/hip_runtime.h>
#include <hip/hip_bf16.h>
#include <stdint.h>

// XCA attention via Gram-matrix factorization:
//   G_b = X_b X_b^T                      (bf16 WMMA, fp32 accum, atomics)
//   Tq = Wq G, Tk = Wk G                 (fp32 VALU, tiny)
//   logits/norms/softmax -> M_b          (fp32 VALU, tiny)
//   out_b = M_b X_b                      (bf16 WMMA, fp32 accum)
// Workspace: G (4.7MB) + T (9.4MB) + M (4.7MB) = ~18.9 MB fp32.
// Both WMMA GEMMs use register-pipelined staging: global loads for step k+1
// issue before the WMMA block of step k, hiding HBM/L2 latency.

typedef __attribute__((ext_vector_type(16))) __bf16 v16bf;
typedef __attribute__((ext_vector_type(8)))  float  v8f;

#define HWN   16384
#define CDIM  384
#define LDST  40   // LDS row stride in halves (32 data + 8 pad), 16B-aligned rows

static __device__ __forceinline__ __bf16 f2bf(float f) {
    union { float f; uint32_t u; } v; v.f = f;
    uint32_t r = v.u + 0x7FFFu + ((v.u >> 16) & 1u);   // round-to-nearest-even
    return __builtin_bit_cast(__bf16, (unsigned short)(r >> 16));
}

// A-matrix 16x32 bf16 fragment (ISA 7.12.2): lanes 0-15 rows M=0..15 hold
// K={0..7,16..23}; lanes 16-31 same rows hold K={8..15,24..31}.
static __device__ __forceinline__ v16bf load_a_frag(const __bf16* base, int rowBase, int lane) {
    int m = lane & 15, g = lane >> 4;
    const __bf16* p0 = base + (rowBase + m) * LDST + 8 * g;
    const __bf16* p1 = p0 + 16;
    v16bf a;
#pragma unroll
    for (int e = 0; e < 8; ++e) { a[e] = p0[e]; a[8 + e] = p1[e]; }
    return a;
}

// B-matrix 32x16 bf16 fragment: lane = column N (mod 16); lanes 0-15 hold
// K={0..7,16..23}, lanes 16-31 hold K={8..15,24..31} (matches A).
// LDS layout Bs[col][k] with k contiguous.
static __device__ __forceinline__ v16bf load_b_frag(const __bf16* base, int colBase, int lane) {
    int n = lane & 15, g = lane >> 4;
    const __bf16* p = base + (colBase + n) * LDST + 8 * g;
    v16bf b;
#pragma unroll
    for (int e = 0; e < 8; ++e) { b[e] = p[e]; b[8 + e] = p[16 + e]; }
    return b;
}

#define WMMA_BF16(A, B, C) \
    __builtin_amdgcn_wmma_f32_16x16x32_bf16(false, (A), false, (B), (short)0, (C), false, false)

__global__ void zero_f32_kernel(float* __restrict__ p, long n) {
    long i = (long)blockIdx.x * blockDim.x + threadIdx.x;
    if (i < n) p[i] = 0.0f;
}

// ---------------------------------------------------------------------------
// Kernel 1: G_b += X_b[c0:c0+128, nchunk] * X_b[cp0:cp0+128, nchunk]^T
// Block: 256 thr = 8 waves (4x2). Block tile 128x128, wave tile 32x64.
// Grid: KSPLIT(8) x 3 x 3 x B(8) = 576 blocks; fp32 atomic accumulation.
// ---------------------------------------------------------------------------
__global__ __launch_bounds__(256) void gram_kernel(const float* __restrict__ x,
                                                   float* __restrict__ G) {
    __shared__ __bf16 As[128 * LDST];
    __shared__ __bf16 Bs[128 * LDST];

    int bx  = blockIdx.x;
    int ks  = bx & 7;  bx >>= 3;
    int cpT = bx % 3;  bx /= 3;
    int cT  = bx % 3;  int b = bx / 3;
    int c0 = cT * 128, cp0 = cpT * 128;
    long xb = (long)b * CDIM * HWN;
    int n0 = ks * 2048;

    int tid = threadIdx.x, lane = tid & 31, wave = tid >> 5;
    int wrow = wave >> 1, wcol = wave & 1;      // 4x2 wave grid

    // Per-thread staging map: 256 channel-rows x 32 n-cols = 2048 float4/step
    const float* srcp[8];
    __bf16*      dstp[8];
#pragma unroll
    for (int t = 0; t < 8; ++t) {
        int idx = tid + t * 256;                // 0..2047
        int row = idx >> 3, seg = idx & 7;
        int ch  = (row < 128) ? (c0 + row) : (cp0 + (row - 128));
        srcp[t] = x + xb + (long)ch * HWN + n0 + seg * 4;
        dstp[t] = ((row < 128) ? (As + row * LDST) : (Bs + (row - 128) * LDST)) + seg * 4;
    }

    v8f acc[2][4];
#pragma unroll
    for (int i = 0; i < 2; ++i)
#pragma unroll
        for (int j = 0; j < 4; ++j)
            acc[i][j] = (v8f){0.f,0.f,0.f,0.f,0.f,0.f,0.f,0.f};

    float4 st[8];
#pragma unroll
    for (int t = 0; t < 8; ++t) st[t] = *(const float4*)(srcp[t]);

    const int NITER = 2048 / 32;                // 64 K-steps of 32
    for (int kk = 0; kk < NITER; ++kk) {
        // convert + store current staging registers to LDS
#pragma unroll
        for (int t = 0; t < 8; ++t) {
            dstp[t][0] = f2bf(st[t].x); dstp[t][1] = f2bf(st[t].y);
            dstp[t][2] = f2bf(st[t].z); dstp[t][3] = f2bf(st[t].w);
        }
        __syncthreads();
        // issue next step's global loads; they overlap the WMMA block below
        if (kk + 1 < NITER) {
#pragma unroll
            for (int t = 0; t < 8; ++t)
                st[t] = *(const float4*)(srcp[t] + (kk + 1) * 32);
        }

        v16bf af[2], bfr[4];
#pragma unroll
        for (int i = 0; i < 2; ++i) af[i]  = load_a_frag(As, wrow * 32 + i * 16, lane);
#pragma unroll
        for (int j = 0; j < 4; ++j) bfr[j] = load_b_frag(Bs, wcol * 64 + j * 16, lane);
#pragma unroll
        for (int i = 0; i < 2; ++i)
#pragma unroll
            for (int j = 0; j < 4; ++j)
                acc[i][j] = WMMA_BF16(af[i], bfr[j], acc[i][j]);
        __syncthreads();
    }

    // C layout: VGPR r -> M = r + 8*(lane>=16), N = lane&15
    int g = lane >> 4, n = lane & 15;
    float* Gb = G + (long)b * CDIM * CDIM;
#pragma unroll
    for (int i = 0; i < 2; ++i)
#pragma unroll
        for (int j = 0; j < 4; ++j)
#pragma unroll
            for (int r = 0; r < 8; ++r) {
                int row = c0  + wrow * 32 + i * 16 + r + 8 * g;
                int col = cp0 + wcol * 64 + j * 16 + n;
                atomicAdd(Gb + (long)row * CDIM + col, acc[i][j][r]);
            }
}

// ---------------------------------------------------------------------------
// Kernel 2a: T[s][b] = W_s * G_b  (s=0:q rows 0..383, s=1:k rows 384..767)
// One block per (b,s,row); 384 threads each own one output column.
// ---------------------------------------------------------------------------
__global__ __launch_bounds__(384) void tmat_kernel(const float* __restrict__ w,
                                                   const float* __restrict__ G,
                                                   float* __restrict__ T) {
    int bx = blockIdx.x;
    int r = bx % CDIM; bx /= CDIM;
    int s = bx & 1;    int b = bx >> 1;
    int j = threadIdx.x;
    const float* wr = w + ((long)s * CDIM + r) * CDIM;     // uniform row -> scalar loads
    const float* Gb = G + (long)b * CDIM * CDIM;
    float sum = 0.0f;
    for (int i = 0; i < CDIM; ++i) sum = fmaf(wr[i], Gb[(long)i * CDIM + j], sum);
    T[((((long)s * 8 + b) * CDIM) + r) * CDIM + j] = sum;
}

// ---------------------------------------------------------------------------
// Kernel 2b: per (b,h): norms, logits, softmax, M = attn o Wv. fp32, tiny.
// ---------------------------------------------------------------------------
__global__ __launch_bounds__(256) void attn_kernel(const float* __restrict__ w,
                                                   const float* __restrict__ T,
                                                   float* __restrict__ M) {
    __shared__ float A[48][48];
    __shared__ float nq[48], nk[48];
    int b = blockIdx.x >> 3, h = blockIdx.x & 7;
    int tid = threadIdx.x;
    const float* Tq = T + (((long)0 * 8 + b) * CDIM + h * 48) * CDIM;
    const float* Tk = T + (((long)1 * 8 + b) * CDIM + h * 48) * CDIM;
    const float* Wq = w + ((long)0 * CDIM + h * 48) * CDIM;
    const float* Wk = w + ((long)1 * CDIM + h * 48) * CDIM;
    const float* Wv = w + ((long)2 * CDIM + h * 48) * CDIM;

    // ||q_c|| = sqrt((Wq G Wq^T)[c,c]) = sqrt(Tq[c]·Wq[c]); same for k.
    for (int idx = tid; idx < 96; idx += 256) {
        int c = idx % 48;
        const float* t  = (idx < 48) ? (Tq + (long)c * CDIM) : (Tk + (long)c * CDIM);
        const float* ww = (idx < 48) ? (Wq + (long)c * CDIM) : (Wk + (long)c * CDIM);
        float s = 0.0f;
        for (int j = 0; j < CDIM; ++j) s = fmaf(t[j], ww[j], s);
        s = fmaxf(sqrtf(fmaxf(s, 0.0f)), 1e-12f);
        if (idx < 48) nq[c] = s; else nk[c] = s;
    }
    __syncthreads();

    // logits: (q_c . k_d) / (nq nk sqrt(hw))
    for (int idx = tid; idx < 48 * 48; idx += 256) {
        int c = idx / 48, d = idx % 48;
        const float* t  = Tq + (long)c * CDIM;
        const float* ww = Wk + (long)d * CDIM;
        float s = 0.0f;
        for (int j = 0; j < CDIM; ++j) s = fmaf(t[j], ww[j], s);
        A[c][d] = s / (nq[c] * nk[d] * 128.0f);   // sqrt(16384) = 128
    }
    __syncthreads();

    if (tid < 48) {
        float mx = -1e30f;
        for (int d = 0; d < 48; ++d) mx = fmaxf(mx, A[tid][d]);
        float sum = 0.0f;
        for (int d = 0; d < 48; ++d) { float e = __expf(A[tid][d] - mx); A[tid][d] = e; sum += e; }
        float inv = 1.0f / sum;
        for (int d = 0; d < 48; ++d) A[tid][d] *= inv;
    }
    __syncthreads();

    // M[h*48+c, :] = sum_d attn[c,d] * Wv[h*48+d, :]
    float* Mb = M + ((long)b * CDIM + h * 48) * CDIM;
    for (int idx = tid; idx < 48 * CDIM; idx += 256) {
        int c = idx / CDIM, j = idx % CDIM;
        float s = 0.0f;
        for (int d = 0; d < 48; ++d) s = fmaf(A[c][d], Wv[(long)d * CDIM + j], s);
        Mb[(long)c * CDIM + j] = s;
    }
}

// ---------------------------------------------------------------------------
// Kernel 3: out_b = M_b (384x384) * X_b (384x16384). Block tile 128(o)x128(n),
// 8 waves (4x2), wave tile 32x64, K loop = 384 in 12 steps of 32.
// ---------------------------------------------------------------------------
__global__ __launch_bounds__(256) void out_gemm_kernel(const float* __restrict__ Mm,
                                                       const float* __restrict__ x,
                                                       float* __restrict__ out) {
    __shared__ __bf16 As[128 * LDST];
    __shared__ __bf16 Bs[128 * LDST];

    int bx = blockIdx.x;
    int nT = bx % 128; bx /= 128;
    int cT = bx % 3;   int b = bx / 3;
    int o0 = cT * 128, n0 = nT * 128;
    long xb = (long)b * CDIM * HWN;
    const float* Mb = Mm + (long)b * CDIM * CDIM;

    int tid = threadIdx.x, lane = tid & 31, wave = tid >> 5;
    int wrow = wave >> 1, wcol = wave & 1;

    // Staging maps.
    // A: M rows o0..o0+127, k chunk (row-major, k contiguous): 1024 float4/step
    const float* srcA[4];
    __bf16*      dstA[4];
#pragma unroll
    for (int t = 0; t < 4; ++t) {
        int idx = tid + t * 256;
        int row = idx >> 3, seg = idx & 7;
        srcA[t] = Mb + (long)(o0 + row) * CDIM + seg * 4;   // +k0 per step
        dstA[t] = As + row * LDST + seg * 4;
    }
    // B: X[k0+k][n0+col] -> Bs[col][k]: 1024 float4/step (32 k x 32 col-quads)
    const float* srcB[4];
    int kB[4], csB[4];
#pragma unroll
    for (int t = 0; t < 4; ++t) {
        int idx = tid + t * 256;
        kB[t] = idx >> 5; csB[t] = idx & 31;
        srcB[t] = x + xb + (long)kB[t] * HWN + n0 + csB[t] * 4;  // +k0*HWN per step
    }

    v8f acc[2][4];
#pragma unroll
    for (int i = 0; i < 2; ++i)
#pragma unroll
        for (int j = 0; j < 4; ++j)
            acc[i][j] = (v8f){0.f,0.f,0.f,0.f,0.f,0.f,0.f,0.f};

    float4 stA[4], stB[4];
#pragma unroll
    for (int t = 0; t < 4; ++t) { stA[t] = *(const float4*)(srcA[t]); stB[t] = *(const float4*)(srcB[t]); }

    const int NITER = CDIM / 32;                 // 12 K-steps
    for (int ki = 0; ki < NITER; ++ki) {
#pragma unroll
        for (int t = 0; t < 4; ++t) {
            dstA[t][0] = f2bf(stA[t].x); dstA[t][1] = f2bf(stA[t].y);
            dstA[t][2] = f2bf(stA[t].z); dstA[t][3] = f2bf(stA[t].w);
            __bf16* db = Bs + csB[t] * 4 * LDST + kB[t];
            db[0 * LDST] = f2bf(stB[t].x);
            db[1 * LDST] = f2bf(stB[t].y);
            db[2 * LDST] = f2bf(stB[t].z);
            db[3 * LDST] = f2bf(stB[t].w);
        }
        __syncthreads();
        if (ki + 1 < NITER) {
#pragma unroll
            for (int t = 0; t < 4; ++t) {
                stA[t] = *(const float4*)(srcA[t] + (ki + 1) * 32);
                stB[t] = *(const float4*)(srcB[t] + (long)(ki + 1) * 32 * HWN);
            }
        }

        v16bf af[2], bfr[4];
#pragma unroll
        for (int i = 0; i < 2; ++i) af[i]  = load_a_frag(As, wrow * 32 + i * 16, lane);
#pragma unroll
        for (int j = 0; j < 4; ++j) bfr[j] = load_b_frag(Bs, wcol * 64 + j * 16, lane);
#pragma unroll
        for (int i = 0; i < 2; ++i)
#pragma unroll
            for (int j = 0; j < 4; ++j)
                acc[i][j] = WMMA_BF16(af[i], bfr[j], acc[i][j]);
        __syncthreads();
    }

    int g = lane >> 4, n = lane & 15;
    float* ob = out + (long)b * CDIM * HWN;
#pragma unroll
    for (int i = 0; i < 2; ++i)
#pragma unroll
        for (int j = 0; j < 4; ++j)
#pragma unroll
            for (int r = 0; r < 8; ++r) {
                int row = o0 + wrow * 32 + i * 16 + r + 8 * g;
                int col = n0 + wcol * 64 + j * 16 + n;
                ob[(long)row * HWN + col] = acc[i][j][r];
            }
}

extern "C" void kernel_launch(void* const* d_in, const int* in_sizes, int n_in,
                              void* d_out, int out_size, void* d_ws, size_t ws_size,
                              hipStream_t stream) {
    const float* x = (const float*)d_in[0];   // [8, 384, 128, 128]
    const float* w = (const float*)d_in[1];   // [1152, 384]
    float* out = (float*)d_out;               // [8, 384, 128, 128]

    float* G = (float*)d_ws;                  //  8*384*384
    float* T = G + (long)8 * CDIM * CDIM;     // 2*8*384*384
    float* M = T + (long)2 * 8 * CDIM * CDIM; //  8*384*384

    long gN = (long)8 * CDIM * CDIM;
    zero_f32_kernel<<<(int)((gN + 255) / 256), 256, 0, stream>>>(G, gN);
    gram_kernel<<<8 * 3 * 3 * 8, 256, 0, stream>>>(x, G);
    tmat_kernel<<<8 * 2 * CDIM, 384, 0, stream>>>(w, G, T);
    attn_kernel<<<64, 256, 0, stream>>>(w, T, M);
    out_gemm_kernel<<<3 * 128 * 8, 256, 0, stream>>>(M, x, out);
}
